// Transformer_82617990906029
// MI455X (gfx1250) — matrix operand order repaired and checked by reading, not compiled, not run on hardware
//
#include <hip/hip_runtime.h>
#include <math.h>

// ---------------------------------------------------------------------------
// CDNA5 (gfx1250, wave32) implementation of a 6-layer DAB-DETR-style decoder.
// All heavy GEMMs use v_wmma_f32_16x16x32_f16 (f16 inputs, f32 accumulate).
// Cross/self attention is flash-style (online softmax) with WMMA for both the
// QK^T and PV GEMMs; K/V tiles are staged once per block cooperatively.
// LDS tiles are stored as packed f16 pairs in fragment order so every WMMA
// operand gather is exactly two ds_load_b128 ops.  All hot global accesses are
// forced to address_space(1) so they lower to global_load/store (not flat).
// Scratch use: ~84 MB of d_ws.   Requires K % 32 == 0 (true for 256/512/2048).
// ---------------------------------------------------------------------------

typedef __attribute__((ext_vector_type(16))) _Float16 v16h;
typedef __attribute__((ext_vector_type(8)))  float    v8f;
typedef __attribute__((ext_vector_type(2)))  float    f32x2;

namespace {
constexpr int kD  = 256;
constexpr int kH  = 8;
constexpr int kNQ = 300;
constexpr int kBS = 8;
constexpr int kHW = 1444;
constexpr int kFF = 2048;
constexpr int kNL = 6;
constexpr int kNQB = kNQ * kBS;   // 2400 rows (q*BS+b)
constexpr int kHWB = kHW * kBS;   // 11552 rows (s*BS+b)
constexpr int kPad = 20;          // padded row stride in packed words (80B, 16B-aligned)
}

union Frag16 { v16h h; uint4 q[2]; };
union Pack2  { uint32_t w; _Float16 h[2]; };

#define GAS __attribute__((address_space(1)))
__device__ __forceinline__ float  ldg1(const float* p) { return *(const GAS float*)p; }
__device__ __forceinline__ f32x2  ldg2(const float* p) { return *(const GAS f32x2*)p; }
__device__ __forceinline__ void   stg1(float* p, float v) { *(GAS float*)p = v; }

// ------------------------------- GEMM --------------------------------------
// C[M,N] = act( A[M,K] @ W[K,N] + bias (+C if accumulate) )
// Block: 256 thr = 8 waves.  Tile 128x64: wave = 32 rows x 32 cols
// (2 A-frags x 2 B-frags -> 4 WMMA per 32-K chunk).
__global__ __launch_bounds__(256) void gemm_f16_wmma_kernel(
    const float* __restrict__ A, const float* __restrict__ W,
    const float* __restrict__ bias, float* __restrict__ C,
    int M, int N, int K, int act, int accumulate)
{
  __shared__ uint32_t As32[128][kPad];   // [row][K-pair]
  __shared__ uint32_t BsT32[64][kPad];   // [col][K-pair]
  const int tid  = threadIdx.x;
  const int wave = tid >> 5;
  const int lane = tid & 31;
  const int ln   = lane & 15;
  const int lh   = lane >> 4;
  const int m0   = blockIdx.y * 128;
  const int n0   = blockIdx.x * 64;
  const int wr   = wave & 3;        // 4 row groups of 32
  const int wc   = wave >> 2;       // 2 col groups of 32
  const int wm   = m0 + wr * 32;
  const int wn   = n0 + wc * 32;

  // Per-thread staging slots fixed across the K loop (branchless staging).
  const float* apk[8]; uint32_t* asd[8]; bool am[8];
  #pragma unroll
  for (int it = 0; it < 8; ++it) {
    const int idx = tid + it * 256;
    const int r = idx >> 4, u = idx & 15;
    const int gm = m0 + r;
    am[it] = gm < M;
    const int gmc = am[it] ? gm : (M - 1);
    apk[it] = A + (size_t)gmc * K + 2 * u;
    asd[it] = &As32[r][u];
  }
  const float* bpk[4]; uint32_t* bsd[4]; bool bm[4];
  #pragma unroll
  for (int it = 0; it < 4; ++it) {
    const int idx = tid + it * 256;
    const int c = idx & 63, u = idx >> 6;
    const int gn = n0 + c;
    bm[it] = gn < N;
    const int gnc = bm[it] ? gn : (N - 1);
    bpk[it] = W + (size_t)(2 * u) * N + gnc;
    bsd[it] = &BsT32[c][u];
  }

  v8f acc00 = {}, acc01 = {}, acc10 = {}, acc11 = {};

  for (int k0 = 0; k0 < K; k0 += 32) {
    // batched staging loads (all in flight before conversion)
    f32x2 av[8]; float bv0[4], bv1[4];
    #pragma unroll
    for (int it = 0; it < 8; ++it) av[it] = ldg2(apk[it]);
    #pragma unroll
    for (int it = 0; it < 4; ++it) { bv0[it] = ldg1(bpk[it]); bv1[it] = ldg1(bpk[it] + N); }
    #pragma unroll
    for (int it = 0; it < 8; ++it) {
      Pack2 pa;
      pa.h[0] = (_Float16)(am[it] ? av[it].x : 0.f);
      pa.h[1] = (_Float16)(am[it] ? av[it].y : 0.f);
      *asd[it] = pa.w;
    }
    #pragma unroll
    for (int it = 0; it < 4; ++it) {
      Pack2 pb;
      pb.h[0] = (_Float16)(bm[it] ? bv0[it] : 0.f);
      pb.h[1] = (_Float16)(bm[it] ? bv1[it] : 0.f);
      *bsd[it] = pb.w;
    }
    if (k0 + 32 < K) {   // global_prefetch_b8 for next K tiles
      __builtin_prefetch(apk[0] + 32, 0, 1);
      __builtin_prefetch(bpk[0] + (size_t)32 * N, 0, 1);
    }
    #pragma unroll
    for (int it = 0; it < 8; ++it) apk[it] += 32;
    #pragma unroll
    for (int it = 0; it < 4; ++it) bpk[it] += (size_t)32 * N;
    __syncthreads();

    // A fragments: rows wr*32+ln and wr*32+16+ln
    Frag16 af0, af1;
    {
      const uint32_t* r0 = &As32[wr * 32 + ln][0];
      const uint32_t* r1 = &As32[wr * 32 + 16 + ln][0];
      af0.q[0] = *(const uint4*)&r0[lh * 8];
      af0.q[1] = *(const uint4*)&r0[lh * 8 + 4];
      af1.q[0] = *(const uint4*)&r1[lh * 8];
      af1.q[1] = *(const uint4*)&r1[lh * 8 + 4];
    }
    // B fragments: cols wc*32+ln and wc*32+16+ln
    Frag16 bf0, bf1;
    {
      const uint32_t* c0p = &BsT32[wc * 32 + ln][0];
      const uint32_t* c1p = &BsT32[wc * 32 + 16 + ln][0];
      bf0.q[0] = *(const uint4*)&c0p[lh * 8];
      bf0.q[1] = *(const uint4*)&c0p[lh * 8 + 4];
      bf1.q[0] = *(const uint4*)&c1p[lh * 8];
      bf1.q[1] = *(const uint4*)&c1p[lh * 8 + 4];
    }
    acc00 = __builtin_amdgcn_wmma_f32_16x16x32_f16(false, af0.h, false, bf0.h, (short)0, acc00, false, false);
    acc01 = __builtin_amdgcn_wmma_f32_16x16x32_f16(false, af0.h, false, bf1.h, (short)0, acc01, false, false);
    acc10 = __builtin_amdgcn_wmma_f32_16x16x32_f16(false, af1.h, false, bf0.h, (short)0, acc10, false, false);
    acc11 = __builtin_amdgcn_wmma_f32_16x16x32_f16(false, af1.h, false, bf1.h, (short)0, acc11, false, false);
    __syncthreads();
  }

  // epilogue: C layout lane -> (row = lh*8+r, col = ln)
  #pragma unroll
  for (int a = 0; a < 2; ++a) {
    #pragma unroll
    for (int t = 0; t < 2; ++t) {
      const v8f acc = (a == 0) ? ((t == 0) ? acc00 : acc01)
                               : ((t == 0) ? acc10 : acc11);
      const int n = wn + t * 16 + ln;
      if (n < N) {
        const float bv = bias ? ldg1(&bias[n]) : 0.f;
        #pragma unroll
        for (int r = 0; r < 8; ++r) {
          const int m = wm + a * 16 + lh * 8 + r;
          if (m < M) {
            float v = acc[r] + bv;
            if (accumulate) v += ldg1(&C[(size_t)m * N + n]);
            if (act == 1) v = fmaxf(v, 0.f);
            stg1(&C[(size_t)m * N + n], v);
          }
        }
      }
    }
  }
}

// ----------------------------- Attention -----------------------------------
__device__ inline float half16_max(float v) {
  #pragma unroll
  for (int off = 8; off; off >>= 1) v = fmaxf(v, __shfl_xor(v, off, 16));
  return v;
}
__device__ inline float half16_sum(float v) {
  #pragma unroll
  for (int off = 8; off; off >>= 1) v += __shfl_xor(v, off, 16);
  return v;
}

// Flash-style multi-head attention, H=8, dv=32, head dim = 32 (Qb==null) or
// 64 = concat(Qa_h, Qb_h) (cross-attn).  Buffers are (rows=seq*B, 256) with
// row = position*B + b, head h at columns [h*32, h*32+32).
// grid = (ceil(L/128), B*H); block = 256 (8 waves, 16 query rows per wave).
__global__ __launch_bounds__(256) void attn_kernel(
    const float* __restrict__ Qa, const float* __restrict__ Qb,
    const float* __restrict__ Ka, const float* __restrict__ Kb,
    const float* __restrict__ V, float* __restrict__ O,
    int L, int S, int B, float scale)
{
  __shared__ uint32_t PlT[8][16 * kPad];  // per-wave P tile, packed K-pairs
  __shared__ uint32_t KaT[32][kPad];      // key tile (cols = s, pairs = feature)
  __shared__ uint32_t KbT[32][kPad];
  __shared__ uint32_t VT[32][kPad];       // value tile (cols = dv, pairs = s)
  const int tid  = threadIdx.x;
  const int wave = tid >> 5;
  const int lane = tid & 31;
  const int ln   = lane & 15;
  const int lh   = lane >> 4;
  const int b    = blockIdx.y >> 3;
  const int h    = blockIdx.y & 7;
  const int qbase = blockIdx.x * 128 + wave * 16;

  // Q as A-fragment(s): lane holds query row (clamped; extra rows never stored)
  v16h qfa; v16h qfb = {};
  {
    const int l = qbase + ln;
    const int lc = (l < L) ? l : (L - 1);
    const size_t qrow = ((size_t)lc * B + b) * kD + (size_t)h * 32;
    #pragma unroll
    for (int p = 0; p < 8; ++p) {
      const int kk = lh * 16 + p * 2;
      const f32x2 fa = ldg2(&Qa[qrow + kk]);
      qfa[2 * p]     = (_Float16)fa.x;
      qfa[2 * p + 1] = (_Float16)fa.y;
      if (Qb) {
        const f32x2 fb = ldg2(&Qb[qrow + kk]);
        qfb[2 * p]     = (_Float16)fb.x;
        qfb[2 * p + 1] = (_Float16)fb.y;
      }
    }
  }

  float rowM[8], rowSum[8];
  #pragma unroll
  for (int r = 0; r < 8; ++r) { rowM[r] = -INFINITY; rowSum[r] = 0.f; }
  v8f o0 = {}; v8f o1 = {};

  auto step = [&](int s0, bool tail) {
    // ---- cooperative staging: K tile (32 s x 32 feat), V tile (32 s x 32 dv)
    #pragma unroll
    for (int it = 0; it < 2; ++it) {
      const int idx = tid + it * 256;
      const int sc = idx >> 4, u = idx & 15;
      const int s = s0 + sc;
      const int scl = (tail && s >= S) ? (S - 1) : s;
      const size_t krow = ((size_t)scl * B + b) * kD + (size_t)h * 32 + 2 * u;
      f32x2 fa = ldg2(&Ka[krow]);
      if (tail && s >= S) { fa.x = 0.f; fa.y = 0.f; }
      Pack2 pa; pa.h[0] = (_Float16)fa.x; pa.h[1] = (_Float16)fa.y;
      KaT[sc][u] = pa.w;
      if (Kb) {
        f32x2 fb = ldg2(&Kb[krow]);
        if (tail && s >= S) { fb.x = 0.f; fb.y = 0.f; }
        Pack2 pb; pb.h[0] = (_Float16)fb.x; pb.h[1] = (_Float16)fb.y;
        KbT[sc][u] = pb.w;
      }
    }
    #pragma unroll
    for (int it = 0; it < 2; ++it) {
      const int idx = tid + it * 256;
      const int d = idx & 31, u = idx >> 5;
      const int s = s0 + 2 * u;
      const int sc0 = (tail && s     >= S) ? (S - 1) : s;
      const int sc1 = (tail && s + 1 >= S) ? (S - 1) : (s + 1);
      const size_t vcol = (size_t)h * 32 + d;
      float f0 = ldg1(&V[((size_t)sc0 * B + b) * kD + vcol]);
      float f1 = ldg1(&V[((size_t)sc1 * B + b) * kD + vcol]);
      if (tail) { if (s >= S) f0 = 0.f; if (s + 1 >= S) f1 = 0.f; }
      Pack2 p; p.h[0] = (_Float16)f0; p.h[1] = (_Float16)f1;
      VT[d][u] = p.w;
    }
    __syncthreads();

    // ---- logits: two 16x16 tiles, fragments straight from LDS ----
    v8f lg0 = {}; v8f lg1 = {};
    #pragma unroll
    for (int t = 0; t < 2; ++t) {
      Frag16 kfa;
      const uint32_t* cp = &KaT[t * 16 + ln][0];
      kfa.q[0] = *(const uint4*)&cp[lh * 8];
      kfa.q[1] = *(const uint4*)&cp[lh * 8 + 4];
      v8f c = {};
      c = __builtin_amdgcn_wmma_f32_16x16x32_f16(false, qfa, false, kfa.h, (short)0, c, false, false);
      if (Qb) {
        Frag16 kfb;
        const uint32_t* cp2 = &KbT[t * 16 + ln][0];
        kfb.q[0] = *(const uint4*)&cp2[lh * 8];
        kfb.q[1] = *(const uint4*)&cp2[lh * 8 + 4];
        c = __builtin_amdgcn_wmma_f32_16x16x32_f16(false, qfb, false, kfb.h, (short)0, c, false, false);
      }
      if (t == 0) lg0 = c; else lg1 = c;
    }
    // V fragments (read before barrier-free P rewrite; kept in registers)
    Frag16 vf0, vf1;
    {
      const uint32_t* v0p = &VT[ln][0];
      const uint32_t* v1p = &VT[16 + ln][0];
      vf0.q[0] = *(const uint4*)&v0p[lh * 8];
      vf0.q[1] = *(const uint4*)&v0p[lh * 8 + 4];
      vf1.q[0] = *(const uint4*)&v1p[lh * 8];
      vf1.q[1] = *(const uint4*)&v1p[lh * 8 + 4];
    }

    // ---- online softmax (row = lh*8+r, replicated over 16 lanes) ----
    const bool m0ok = !tail || (s0 + ln) < S;
    const bool m1ok = !tail || (s0 + 16 + ln) < S;
    _Float16* ph = (_Float16*)&PlT[wave][0];   // row stride 2*kPad halves
    #pragma unroll
    for (int r = 0; r < 8; ++r) {
      const float l0 = m0ok ? lg0[r] * scale : -INFINITY;
      const float l1 = m1ok ? lg1[r] * scale : -INFINITY;
      const float nm   = half16_max(fmaxf(l0, l1));
      const float newM = fmaxf(rowM[r], nm);
      const float f    = __expf(rowM[r] - newM);   // exp(-inf)=0 on first block
      rowM[r] = newM;
      const float p0 = __expf(l0 - newM);
      const float p1 = __expf(l1 - newM);
      rowSum[r] = rowSum[r] * f + half16_sum(p0 + p1);
      o0[r] *= f; o1[r] *= f;
      ph[(lh * 8 + r) * (2 * kPad) + ln]      = (_Float16)p0;
      ph[(lh * 8 + r) * (2 * kPad) + 16 + ln] = (_Float16)p1;
    }
    // P as A-fragment.  PlT[wave] is wave-private and LDS ops from one wave
    // complete in order (ISA 7.1), so no block barrier is needed here.
    Frag16 pf;
    {
      const uint32_t* rp = &PlT[wave][ln * kPad];
      pf.q[0] = *(const uint4*)&rp[lh * 8];
      pf.q[1] = *(const uint4*)&rp[lh * 8 + 4];
    }
    o0 = __builtin_amdgcn_wmma_f32_16x16x32_f16(false, pf.h, false, vf0.h, (short)0, o0, false, false);
    o1 = __builtin_amdgcn_wmma_f32_16x16x32_f16(false, pf.h, false, vf1.h, (short)0, o1, false, false);
    __syncthreads();   // K/V tiles consumed; safe to restage
  };

  int s0 = 0;
  for (; s0 + 32 <= S; s0 += 32) step(s0, false);
  if (s0 < S) step(s0, true);

  #pragma unroll
  for (int r = 0; r < 8; ++r) {
    const int l = qbase + lh * 8 + r;
    if (l < L) {
      const float inv = 1.f / rowSum[r];
      const size_t orow = ((size_t)l * B + b) * kD + (size_t)h * 32;
      stg1((float*)&O[orow + ln],      o0[r] * inv);
      stg1((float*)&O[orow + 16 + ln], o1[r] * inv);
    }
  }
}

// ------------------------- elementwise / LN kernels ------------------------
__device__ inline float block256_sum(float v, float* red8, int tid) {
  #pragma unroll
  for (int off = 16; off; off >>= 1) v += __shfl_xor(v, off, 32);
  if ((tid & 31) == 0) red8[tid >> 5] = v;
  __syncthreads();
  float s = 0.f;
  #pragma unroll
  for (int i = 0; i < 8; ++i) s += red8[i];
  __syncthreads();
  return s;
}

__global__ __launch_bounds__(256) void resln_kernel(
    const float* __restrict__ x, const float* __restrict__ delta,
    const float* __restrict__ g, const float* __restrict__ be,
    float* __restrict__ out)
{
  __shared__ float red8[8];
  const int row = blockIdx.x;
  const int d = threadIdx.x;
  float v = ldg1(&x[(size_t)row * kD + d]);
  if (delta) v += ldg1(&delta[(size_t)row * kD + d]);
  const float mean = block256_sum(v, red8, d) * (1.f / kD);
  const float c = v - mean;
  const float var = block256_sum(c * c, red8, d) * (1.f / kD);
  stg1(&out[(size_t)row * kD + d], c * rsqrtf(var + 1e-5f) * ldg1(&g[d]) + ldg1(&be[d]));
}

// LN(x) written transposed into one hs slot: (q*B+b, d) -> (b, q, d)
__global__ __launch_bounds__(256) void ln_to_hs_kernel(
    const float* __restrict__ x, const float* __restrict__ g,
    const float* __restrict__ be, float* __restrict__ hs_slot, int B, int NQl)
{
  __shared__ float red8[8];
  const int row = blockIdx.x;
  const int d = threadIdx.x;
  const float v = ldg1(&x[(size_t)row * kD + d]);
  const float mean = block256_sum(v, red8, d) * (1.f / kD);
  const float c = v - mean;
  const float var = block256_sum(c * c, red8, d) * (1.f / kD);
  const int q = row / B, b = row % B;
  stg1(&hs_slot[((size_t)b * NQl + q) * kD + d],
       c * rsqrtf(var + 1e-5f) * ldg1(&g[d]) + ldg1(&be[d]));
}

__global__ void init_a_kernel(const float* __restrict__ ref, float* __restrict__ a,
                              float* __restrict__ rf0, int rows, int B, int NQl)
{
  const int idx = blockIdx.x * 256 + threadIdx.x;
  if (idx >= rows * 4) return;
  const float v = 1.f / (1.f + __expf(-ldg1(&ref[idx])));
  stg1(&a[idx], v);
  const int row = idx >> 2, c = idx & 3;
  const int q = row / B, b = row % B;
  stg1(&rf0[((size_t)b * NQl + q) * 4 + c], v);
}

// qse_full = concat(emb(a1), emb(a0), emb(a2), emb(a3)), each 128 dims:
// out[2m]=sin(x*2pi/t_m), out[2m+1]=cos(x*2pi/t_m), t_m=10000^(m/64)
__global__ void sine_embed_kernel(const float* __restrict__ a,
                                  float* __restrict__ qse_full, int rows)
{
  const int idx = blockIdx.x * 256 + threadIdx.x;
  if (idx >= rows * 512) return;
  const int row = idx >> 9;
  const int j = idx & 511;
  const int part = j >> 7;
  const int coord[4] = {1, 0, 2, 3};
  const int jj = j & 127;
  const int m = jj >> 1;
  const float t = __powf(10000.f, (float)m * (1.f / 64.f));
  const float p = ldg1(&a[row * 4 + coord[part]]) * (6.283185307179586f) / t;
  stg1(&qse_full[idx], (jj & 1) ? __cosf(p) : __sinf(p));
}

// qse = qse_full[:,:256] * (qscale?) then halves scaled by refhw/a ratios
__global__ void qse_prep_kernel(const float* __restrict__ qse_full,
                                const float* __restrict__ qscale,
                                const float* __restrict__ ra_raw,
                                const float* __restrict__ a,
                                float* __restrict__ qse, int rows)
{
  const int idx = blockIdx.x * 256 + threadIdx.x;
  if (idx >= rows * kD) return;
  const int row = idx >> 8;
  const int d = idx & 255;
  float v = ldg1(&qse_full[(size_t)row * 512 + d]);
  if (qscale) v *= ldg1(&qscale[idx]);
  const int lo = (d < 128);
  const float rh = 1.f / (1.f + __expf(-ldg1(&ra_raw[row * 2 + (lo ? 1 : 0)])));
  const float av = ldg1(&a[row * 4 + (lo ? 3 : 2)]);
  stg1(&qse[idx], v * (rh / av));
}

// b_pred = sigmoid(inv_sig(a) + db); write transposed rf slot; a <- b_pred
__global__ void ref_update_kernel(float* __restrict__ a, const float* __restrict__ db,
                                  float* __restrict__ rf_slot, int rows, int B, int NQl)
{
  const int idx = blockIdx.x * 256 + threadIdx.x;
  if (idx >= rows * 4) return;
  float av = ldg1(&a[idx]);
  av = fminf(fmaxf(av, 1e-3f), 1.f - 1e-3f);
  const float is = logf(av / (1.f - av));
  const float bp = 1.f / (1.f + __expf(-(is + ldg1(&db[idx]))));
  stg1(&a[idx], bp);
  const int row = idx >> 2, c = idx & 3;
  const int q = row / B, b = row % B;
  stg1(&rf_slot[((size_t)b * NQl + q) * 4 + c], bp);
}

// ------------------------------- host side ---------------------------------
extern "C" void kernel_launch(void* const* d_in, const int* in_sizes, int n_in,
                              void* d_out, int out_size, void* d_ws, size_t ws_size,
                              hipStream_t stream)
{
  (void)in_sizes; (void)n_in; (void)out_size; (void)ws_size;
  // Leaf order assumes jax-tree flattening (dict keys sorted recursively):
  // 0: memory
  // params: 1-3 bb_b[0..2], 4-6 bb_w[0..2], 7..210 layers (6 x 34 sorted keys),
  //         211 norm_b, 212 norm_g, 213-214 qs_b, 215-216 qs_w,
  //         217-218 ra_b, 219-220 ra_w, 221-222 ref_b, 223-224 ref_w
  // 225: pos, 226: reference_unsigmoid, 227: tgt
  auto F = [&](int i) -> const float* { return (const float*)d_in[i]; };
  const float* memory = F(0);
  const float* bb_b[3] = {F(1), F(2), F(3)};
  const float* bb_w[3] = {F(4), F(5), F(6)};
  const float* norm_b = F(211);
  const float* norm_g = F(212);
  const float* qs_b[2] = {F(213), F(214)};
  const float* qs_w[2] = {F(215), F(216)};
  const float* ra_b[2] = {F(217), F(218)};
  const float* ra_w[2] = {F(219), F(220)};
  const float* ref_b[2] = {F(221), F(222)};
  const float* ref_w[2] = {F(223), F(224)};
  const float* pos = F(225);
  const float* ref_unsig = F(226);
  const float* tgt = F(227);
  // per-layer sorted-key offsets inside a layer dict (34 leaves):
  // 0 ca_kc_b 1 ca_kc_w 2 ca_kp_b 3 ca_kp_w 4 ca_o_b 5 ca_o_w 6 ca_qc_b 7 ca_qc_w
  // 8 ca_qp_b 9 ca_qp_w 10 ca_v_b 11 ca_v_w 12 ff1_b 13 ff1_w 14 ff2_b 15 ff2_w
  // 16 n1_b 17 n1_g 18 n2_b 19 n2_g 20 n3_b 21 n3_g 22 sa_kc_b 23 sa_kc_w
  // 24 sa_kp_b 25 sa_kp_w 26 sa_o_b 27 sa_o_w 28 sa_qc_b 29 sa_qc_w
  // 30 sa_qp_b 31 sa_qp_w 32 sa_v_b 33 sa_v_w

  float* hs = (float*)d_out;                              // (6,BS,NQ,D)
  float* rf = hs + (size_t)kNL * kBS * kNQ * kD;          // (7,BS,NQ,4)

  float* w = (float*)d_ws;
  size_t off = 0;
  auto alloc = [&](size_t n) { float* p = w + off; off += n; return p; };
  float* x    = alloc((size_t)kNQB * kD);
  float* qb   = alloc((size_t)kNQB * kD);
  float* kb   = alloc((size_t)kNQB * kD);
  float* vb   = alloc((size_t)kNQB * kD);
  float* ob   = alloc((size_t)kNQB * kD);
  float* ao   = alloc((size_t)kNQB * kD);
  float* qsef = alloc((size_t)kNQB * 2 * kD);
  float* qpos = alloc((size_t)kNQB * kD);
  float* qse  = alloc((size_t)kNQB * kD);
  float* qsc  = alloc((size_t)kNQB * kD);
  float* h2   = alloc((size_t)kNQB * kD);
  float* hid  = alloc((size_t)kNQB * kFF);
  float* ra_o = alloc((size_t)kNQB * 2);
  float* db   = alloc((size_t)kNQB * 4);
  float* ab   = alloc((size_t)kNQB * 4);
  float* kc_m = alloc((size_t)kHWB * kD);
  float* kp_m = alloc((size_t)kHWB * kD);
  float* v_m  = alloc((size_t)kHWB * kD);

  auto gemm = [&](const float* A, const float* W_, const float* bias, float* C,
                  int M, int N, int K, int act, int accum) {
    dim3 g((N + 63) / 64, (M + 127) / 128);
    gemm_f16_wmma_kernel<<<g, 256, 0, stream>>>(A, W_, bias, C, M, N, K, act, accum);
  };

  (void)hipMemcpyAsync(x, tgt, (size_t)kNQB * kD * sizeof(float),
                       hipMemcpyDeviceToDevice, stream);
  init_a_kernel<<<(kNQB * 4 + 255) / 256, 256, 0, stream>>>(ref_unsig, ab, rf, kNQB, kBS, kNQ);

  const float sscale = 0.17677669529663687f;  // 1/sqrt(32)
  const float cscale = 0.125f;                // 1/sqrt(64)
  const dim3 attn_grid((kNQ + 127) / 128, kBS * kH);

  for (int i = 0; i < kNL; ++i) {
    const int P = 7 + i * 34;
    auto LPW = [&](int j) { return F(P + j); };

    sine_embed_kernel<<<(kNQB * 512 + 255) / 256, 256, 0, stream>>>(ab, qsef, kNQB);
    // qpos = MLP_ref(qse_full):  512 -> 256 (relu) -> 256
    gemm(qsef, ref_w[0], ref_b[0], hid, kNQB, kD, 2 * kD, 1, 0);
    gemm(hid, ref_w[1], ref_b[1], qpos, kNQB, kD, kD, 0, 0);
    if (i > 0) {  // query scale MLP
      gemm(x, qs_w[0], qs_b[0], hid, kNQB, kD, kD, 1, 0);
      gemm(hid, qs_w[1], qs_b[1], qsc, kNQB, kD, kD, 0, 0);
    }
    // ref anchor h/w head (sigmoid applied in qse_prep)
    gemm(x, ra_w[0], ra_b[0], hid, kNQB, kD, kD, 1, 0);
    gemm(hid, ra_w[1], ra_b[1], ra_o, kNQB, 2, kD, 0, 0);
    qse_prep_kernel<<<(kNQB * kD + 255) / 256, 256, 0, stream>>>(
        qsef, (i > 0) ? qsc : nullptr, ra_o, ab, qse, kNQB);

    // ---- self attention ----
    gemm(x,    LPW(29), LPW(28), qb, kNQB, kD, kD, 0, 0);
    gemm(qpos, LPW(31), LPW(30), qb, kNQB, kD, kD, 0, 1);
    gemm(x,    LPW(23), LPW(22), kb, kNQB, kD, kD, 0, 0);
    gemm(qpos, LPW(25), LPW(24), kb, kNQB, kD, kD, 0, 1);
    gemm(x,    LPW(33), LPW(32), vb, kNQB, kD, kD, 0, 0);
    attn_kernel<<<attn_grid, 256, 0, stream>>>(qb, nullptr, kb, nullptr, vb, ao,
                                               kNQ, kNQ, kBS, sscale);
    gemm(ao, LPW(27), LPW(26), ob, kNQB, kD, kD, 0, 0);
    resln_kernel<<<kNQB, 256, 0, stream>>>(x, ob, LPW(17), LPW(16), x);

    // ---- cross attention (head dim 64 = qc_h || qp_h) ----
    gemm(x,   LPW(7), LPW(6), qb, kNQB, kD, kD, 0, 0);
    gemm(qse, LPW(9), LPW(8), ob, kNQB, kD, kD, 0, 0);          // qp part
    gemm(memory, LPW(1),  LPW(0),  kc_m, kHWB, kD, kD, 0, 0);
    gemm(pos,    LPW(3),  LPW(2),  kp_m, kHWB, kD, kD, 0, 0);
    gemm(memory, LPW(11), LPW(10), v_m,  kHWB, kD, kD, 0, 0);
    attn_kernel<<<attn_grid, 256, 0, stream>>>(qb, ob, kc_m, kp_m, v_m, ao,
                                               kNQ, kHW, kBS, cscale);
    gemm(ao, LPW(5), LPW(4), vb, kNQB, kD, kD, 0, 0);
    resln_kernel<<<kNQB, 256, 0, stream>>>(x, vb, LPW(19), LPW(18), x);

    // ---- FFN ----
    gemm(x, LPW(13), LPW(12), hid, kNQB, kFF, kD, 1, 0);
    gemm(hid, LPW(15), LPW(14), ob, kNQB, kD, kFF, 0, 0);
    resln_kernel<<<kNQB, 256, 0, stream>>>(x, ob, LPW(21), LPW(20), x);

    // ---- box refinement: db = MLP_bb(x), anchors update ----
    gemm(x, bb_w[0], bb_b[0], hid, kNQB, kD, kD, 1, 0);
    gemm(hid, bb_w[1], bb_b[1], h2, kNQB, kD, kD, 1, 0);
    gemm(h2, bb_w[2], bb_b[2], db, kNQB, 4, kD, 0, 0);
    ref_update_kernel<<<(kNQB * 4 + 255) / 256, 256, 0, stream>>>(
        ab, db, rf + (size_t)(i + 1) * kBS * kNQ * 4, kNQB, kBS, kNQ);

    // ---- intermediate normalized output -> hs[i] (transposed) ----
    ln_to_hs_kernel<<<kNQB, 256, 0, stream>>>(
        x, norm_g, norm_b, hs + (size_t)i * kBS * kNQ * kD, kBS, kNQ);
  }
}
